// RansGinoGridToMeshOg_49744311222693
// MI455X (gfx1250) — matrix operand order repaired and checked by reading, not compiled, not run on hardware
//
#include <hip/hip_runtime.h>
#include <math.h>

// ---------------- problem constants ----------------
#define NQ     131072
#define E_EDGE 262144
#define IN_DIM 256
#define PE_DIM 512
#define F0     768    // IN_DIM + PE_DIM
#define H0     512
#define H1     256
#define BOT    128
#define P0N    256
#define OUTN   4

#define MTILE  64     // edges / queries per block (4 row-blocks of 16)

typedef __attribute__((ext_vector_type(16))) __bf16 v16bf;
typedef __attribute__((ext_vector_type(8)))  __bf16 v8bf;
typedef __attribute__((ext_vector_type(8)))  float  v8f;

// ---------------- helpers ----------------
__device__ __forceinline__ float gelu_exact(float x) {
    return 0.5f * x * (1.0f + erff(x * 0.70710678118654752f));
}

// RB row-blocks (16 rows each) of a 16-wide output column tile.
// Each kt loads ONE B fragment (32B/lane) and reuses it for RB wmmas.
template <int RB>
__device__ __forceinline__ void wmma_col_tile(const __bf16* __restrict__ sA, int rowStride,
                                              const __bf16* __restrict__ wFrag,
                                              int K, int nTile, int NT,
                                              float biasVal, int lane, v8f acc[RB]) {
    #pragma unroll
    for (int rb = 0; rb < RB; ++rb)
        #pragma unroll
        for (int r = 0; r < 8; ++r) acc[rb][r] = biasVal;

    const int m   = lane & 15;
    const int aHi = (lane >> 4) << 3;                 // 0 or 8
    const int KT  = K >> 5;
    for (int kt = 0; kt < KT; ++kt) {
        const int kbase = kt << 5;
        const v16bf b = *reinterpret_cast<const v16bf*>(
            wFrag + (((size_t)kt * NT + nTile) * 32 + lane) * 16);
        #pragma unroll
        for (int rb = 0; rb < RB; ++rb) {
            const __bf16* rowPtr = sA + (size_t)(rb * 16 + m) * rowStride;
            v8bf lo = *reinterpret_cast<const v8bf*>(rowPtr + kbase + aHi);
            v8bf hi = *reinterpret_cast<const v8bf*>(rowPtr + kbase + 16 + aHi);
            v16bf a = __builtin_shufflevector(lo, hi,
                        0,1,2,3,4,5,6,7,8,9,10,11,12,13,14,15);
            acc[rb] = __builtin_amdgcn_wmma_f32_16x16x32_bf16(false, a, false, b,
                                                              (short)0, acc[rb], false, false);
        }
    }
}

template <int RB>
__device__ __forceinline__ void store_col_tile(const v8f acc[RB], __bf16* __restrict__ sOut,
                                               int rowStride, int n0, int lane, bool doGelu) {
    const int n     = n0 + (lane & 15);
    const int rbase = (lane >> 4) << 3;
    #pragma unroll
    for (int rb = 0; rb < RB; ++rb) {
        #pragma unroll
        for (int r = 0; r < 8; ++r) {
            float v = acc[rb][r];
            if (doGelu) v = gelu_exact(v);
            sOut[(size_t)(rb * 16 + rbase + r) * rowStride + n] = (__bf16)v;
        }
    }
}

// ---------------- prep kernels ----------------

// zero sums + counts (graph replays re-run this: deterministic)
__global__ void kZero(float* __restrict__ sums, float* __restrict__ counts) {
    size_t i = (size_t)blockIdx.x * blockDim.x + threadIdx.x;
    size_t total = (size_t)NQ * BOT;
    for (size_t k = i; k < total; k += (size_t)gridDim.x * blockDim.x) sums[k] = 0.0f;
    for (size_t k = i; k < NQ; k += (size_t)gridDim.x * blockDim.x) counts[k] = 0.0f;
}

// sincos positional embedding table, bf16: pe[q][c*128 + j]
__global__ void kPE(const float* __restrict__ query_pos, __bf16* __restrict__ peB) {
    const int q = blockIdx.x;
    const float LN1E4_OVER = 9.210340371976184f / 128.0f;  // ln(10000)/eff
    for (int e = threadIdx.x; e < PE_DIM; e += blockDim.x) {
        const int c = e >> 7;
        const int j = e & 127;
        float coord = (c < 3) ? (query_pos[q * 3 + c] * 0.01f - 1.0f) : 1.0f;
        const int jj = (j < 64) ? j : (j - 64);
        float om = __expf(-(float)(2 * jj) * LN1E4_OVER);
        float v  = (j < 64) ? __sinf(coord * om) : __cosf(coord * om);
        peB[(size_t)q * PE_DIM + e] = (__bf16)v;
    }
}

// transform row-major f32 W[K][N] into fragment-ordered bf16 for wmma B operand
__global__ void kFrag(const float* __restrict__ W, __bf16* __restrict__ dst, int K, int N) {
    const int NT = N >> 4;
    const size_t total = (size_t)K * N;
    for (size_t idx = (size_t)blockIdx.x * blockDim.x + threadIdx.x; idx < total;
         idx += (size_t)gridDim.x * blockDim.x) {
        const int j    = (int)(idx & 15);
        const int lane = (int)((idx >> 4) & 31);
        const size_t tile = idx >> 9;
        const int nt = (int)(tile % NT);
        const int kt = (int)(tile / NT);
        const int k = (kt << 5) + ((j >> 3) << 4) + ((lane >> 4) << 3) + (j & 7);  // lanes 0-15: K 0..7,16..23; 16-31: K 8..15,24..31
        const int n = (nt << 4) + (lane & 15);
        dst[idx] = (__bf16)W[(size_t)k * N + n];
    }
}

// ---------------- edge MLP + segment scatter ----------------
// 64 edges per block; 160KB dynamic LDS (sA 64x768 bf16 | sB 64x512 bf16).
__global__ void __launch_bounds__(256)
kEdges(const float* __restrict__ xf, const __bf16* __restrict__ peB,
       const int* __restrict__ edges,
       const __bf16* __restrict__ wf0, const float* __restrict__ b0,
       const __bf16* __restrict__ wf1, const float* __restrict__ b1,
       const __bf16* __restrict__ wf2, const float* __restrict__ b2,
       float* __restrict__ sums, float* __restrict__ counts) {
    extern __shared__ __align__(32) char smem[];
    __bf16* sA = (__bf16*)smem;                       // MTILE x F0 (feat, later layer1 out)
    __bf16* sB = sA + (size_t)MTILE * F0;             // MTILE x H0 (layer0 out)
    __shared__ int sQ[MTILE], sG[MTILE];

    const int tid  = threadIdx.x;
    const int lane = tid & 31;
    const int wv   = tid >> 5;           // 8 waves
    const int e0   = blockIdx.x * MTILE;

    if (tid < MTILE) {
        sQ[tid] = edges[2 * (e0 + tid) + 0];
        sG[tid] = edges[2 * (e0 + tid) + 1];
    }
    __syncthreads();

    if (tid < MTILE) atomicAdd(&counts[sQ[tid]], 1.0f);

    // stage feat = [x-gather (256 f32->bf16) | pe-gather (512 bf16)]
    for (int i = tid; i < MTILE * IN_DIM; i += 256) {
        const int m = i >> 8, c = i & 255;
        sA[(size_t)m * F0 + c] = (__bf16)xf[(size_t)sG[m] * IN_DIM + c];
    }
    {   // pe rows copied as 32-bit chunks (2 bf16 each)
        const unsigned int* peU = (const unsigned int*)peB;
        for (int i = tid; i < MTILE * (PE_DIM / 2); i += 256) {
            const int m = i >> 8, c = i & 255;   // 256 uints per row
            ((unsigned int*)(sA + (size_t)m * F0 + IN_DIM))[c] =
                peU[(size_t)sQ[m] * (PE_DIM / 2) + c];
        }
    }
    __syncthreads();

    // layer 0: 768 -> 512, gelu. 32 n-tiles, 4 per wave, 4 row-blocks each.
    #pragma unroll
    for (int t = 0; t < 4; ++t) {
        const int nt = wv * 4 + t;
        v8f acc[4];
        wmma_col_tile<4>(sA, F0, wf0, F0, nt, H0 / 16,
                         b0[nt * 16 + (lane & 15)], lane, acc);
        store_col_tile<4>(acc, sB, H0, nt * 16, lane, true);
    }
    __syncthreads();

    // layer 1: 512 -> 256, gelu. 16 n-tiles, 2 per wave. Output reuses sA[:, :256].
    #pragma unroll
    for (int t = 0; t < 2; ++t) {
        const int nt = wv * 2 + t;
        v8f acc[4];
        wmma_col_tile<4>(sB, H0, wf1, H0, nt, H1 / 16,
                         b1[nt * 16 + (lane & 15)], lane, acc);
        store_col_tile<4>(acc, sA, F0, nt * 16, lane, true);
    }
    __syncthreads();

    // layer 2: 256 -> 128, no gelu, scatter-add into segment sums.
    {
        const int nt = wv;  // 8 n-tiles, 1 per wave
        v8f acc[4];
        wmma_col_tile<4>(sA, F0, wf2, H1, nt, BOT / 16,
                         b2[nt * 16 + (lane & 15)], lane, acc);
        const int n     = nt * 16 + (lane & 15);
        const int rbase = (lane >> 4) << 3;
        #pragma unroll
        for (int rb = 0; rb < 4; ++rb)
            #pragma unroll
            for (int r = 0; r < 8; ++r) {
                const int q = sQ[rb * 16 + rbase + r];
                atomicAdd(&sums[(size_t)q * BOT + n], acc[rb][r]);
            }
    }
}

// ---------------- prediction head ----------------
// 64 queries per block; 48KB static LDS.
__global__ void __launch_bounds__(256)
kPred(const float* __restrict__ sums, const float* __restrict__ counts,
      const __bf16* __restrict__ pf0, const float* __restrict__ pb0,
      const float* __restrict__ pw1, const float* __restrict__ pb1,
      float* __restrict__ out) {
    __shared__ __attribute__((aligned(32))) __bf16 sM[MTILE][BOT];
    __shared__ __attribute__((aligned(32))) __bf16 sP[MTILE][P0N];

    const int tid  = threadIdx.x;
    const int lane = tid & 31;
    const int wv   = tid >> 5;
    const int q0   = blockIdx.x * MTILE;

    for (int i = tid; i < MTILE * BOT; i += 256) {
        const int m = i >> 7, c = i & 127;
        const int q = q0 + m;
        float cnt = counts[q];
        if (cnt < 1.0f) cnt = 1.0f;
        sM[m][c] = (__bf16)(sums[(size_t)q * BOT + c] / cnt);
    }
    __syncthreads();

    // p0: 128 -> 256, gelu. 16 n-tiles, 2 per wave, 4 row-blocks each.
    #pragma unroll
    for (int t = 0; t < 2; ++t) {
        const int nt = wv * 2 + t;
        v8f acc[4];
        wmma_col_tile<4>(&sM[0][0], BOT, pf0, BOT, nt, P0N / 16,
                         pb0[nt * 16 + (lane & 15)], lane, acc);
        store_col_tile<4>(acc, &sP[0][0], P0N, nt * 16, lane, true);
    }
    __syncthreads();

    // p1: 256 -> 4, tiny -> one VALU dot per thread (64 q x 4 out = 256).
    {
        const int q = tid >> 2, o = tid & 3;
        float acc = pb1[o];
        for (int k = 0; k < P0N; ++k)
            acc += (float)sP[q][k] * pw1[k * OUTN + o];
        out[(size_t)(q0 + q) * OUTN + o] = acc;
    }
}

// ---------------- launch ----------------
extern "C" void kernel_launch(void* const* d_in, const int* in_sizes, int n_in,
                              void* d_out, int out_size, void* d_ws, size_t ws_size,
                              hipStream_t stream) {
    (void)in_sizes; (void)n_in; (void)out_size; (void)ws_size;

    const float* xf     = (const float*)d_in[0];   // (SEQ, 256)
    const float* qpos   = (const float*)d_in[1];   // (NQ, 3)
    const int*   edges  = (const int*)  d_in[2];   // (E, 2)
    const float* w0     = (const float*)d_in[3];   // 768x512
    const float* b0     = (const float*)d_in[4];
    const float* w1     = (const float*)d_in[5];   // 512x256
    const float* b1     = (const float*)d_in[6];
    const float* w2     = (const float*)d_in[7];   // 256x128
    const float* b2     = (const float*)d_in[8];
    const float* pw0    = (const float*)d_in[9];   // 128x256
    const float* pb0    = (const float*)d_in[10];
    const float* pw1    = (const float*)d_in[11];  // 256x4
    const float* pb1    = (const float*)d_in[12];
    float* out          = (float*)d_out;

    // workspace carve (256B aligned)
    char* base = (char*)d_ws;
    size_t off = 0;
    auto carve = [&](size_t bytes) -> char* {
        off = (off + 255) & ~(size_t)255;
        char* p = base + off;
        off += bytes;
        return p;
    };
    __bf16* peB  = (__bf16*)carve((size_t)NQ * PE_DIM * sizeof(__bf16));
    __bf16* wf0  = (__bf16*)carve((size_t)F0  * H0  * sizeof(__bf16));
    __bf16* wf1  = (__bf16*)carve((size_t)H0  * H1  * sizeof(__bf16));
    __bf16* wf2  = (__bf16*)carve((size_t)H1  * BOT * sizeof(__bf16));
    __bf16* pf0  = (__bf16*)carve((size_t)BOT * P0N * sizeof(__bf16));
    float*  sums = (float*) carve((size_t)NQ * BOT * sizeof(float));
    float*  cnts = (float*) carve((size_t)NQ * sizeof(float));

    kZero<<<4096, 256, 0, stream>>>(sums, cnts);
    kPE<<<NQ, 256, 0, stream>>>(qpos, peB);
    kFrag<<<(F0 * H0 + 255) / 256, 256, 0, stream>>>(w0,  wf0, F0,  H0);
    kFrag<<<(H0 * H1 + 255) / 256, 256, 0, stream>>>(w1,  wf1, H0,  H1);
    kFrag<<<(H1 * BOT + 255) / 256, 256, 0, stream>>>(w2,  wf2, H1,  BOT);
    kFrag<<<(BOT * P0N + 255) / 256, 256, 0, stream>>>(pw0, pf0, BOT, P0N);

    const size_t edgeSmem = (size_t)(MTILE * F0 + MTILE * H0) * sizeof(__bf16); // 160KB
    kEdges<<<E_EDGE / MTILE, 256, edgeSmem, stream>>>(xf, peB, edges,
                                                      wf0, b0, wf1, b1, wf2, b2,
                                                      sums, cnts);
    kPred<<<NQ / MTILE, 256, 0, stream>>>(sums, cnts, pf0, pb0, pw1, pb1, out);
}
